// GlymphaticGatedCrossAttention_52055003627698
// MI455X (gfx1250) — compile-verified, matrix operand order checked by reading, not run on hardware
//
#include <hip/hip_runtime.h>
#include <hip/hip_bf16.h>
#include <stdint.h>

#define N_S 2048
#define N_C 4096
#define SLEEP_DIM 1024
#define BRAIN_DIM 512
#define N_HEADS 8
#define D_K 128

typedef __attribute__((ext_vector_type(16))) __bf16 bf16x16;
typedef __attribute__((ext_vector_type(8)))  float  v8f;

union Frag {
  bf16x16 v;
  uint4   q[2];
};

__device__ __forceinline__ uint16_t f32_to_bf16(float x) {
  union { float f; uint32_t u; } a; a.f = x;
  uint32_t r = a.u + 0x7FFFu + ((a.u >> 16) & 1u);   // round-to-nearest-even
  return (uint16_t)(r >> 16);
}

// Load a 16x32 bf16 fragment (A-layout == B-layout by symmetry):
// lane<16: rows row0+lane, K = k0+{0..7} and k0+{16..23}
// lane>=16: rows row0+lane-16, K = k0+{8..15} and k0+{24..31}
__device__ __forceinline__ bf16x16 load_frag_g(const uint16_t* __restrict__ p,
                                               int ld, int row0, int k0) {
  const int lane = threadIdx.x & 31;
  const int r = row0 + (lane & 15);
  const int k = k0 + ((lane >> 4) << 3);
  const uint16_t* base = p + (size_t)r * (size_t)ld + k;
  Frag f;
  f.q[0] = *reinterpret_cast<const uint4*>(base);
  f.q[1] = *reinterpret_cast<const uint4*>(base + 16);
  return f.v;
}

__device__ __forceinline__ bf16x16 load_frag_lds(const uint16_t* p, int ld) {
  const int lane = threadIdx.x & 31;
  const uint16_t* base = p + (lane & 15) * ld + ((lane >> 4) << 3);
  Frag f;
  f.q[0] = *reinterpret_cast<const uint4*>(base);
  f.q[1] = *reinterpret_cast<const uint4*>(base + 16);
  return f.v;
}

// ---------------------------------------------------------------- convert
__global__ void cvt_f32_bf16_kernel(const float* __restrict__ in,
                                    uint16_t* __restrict__ out, int n) {
  int i = blockIdx.x * blockDim.x + threadIdx.x;
  int stride = gridDim.x * blockDim.x;
  for (; i < n; i += stride) out[i] = f32_to_bf16(in[i]);
}

// ---------------------------------------------------------------- NT GEMM
// C[M,N] = A[M,K] (bf16, row-major) * B[N,K]^T (bf16, row-major)
// Each wave computes a 16x64 strip of C via 4 WMMA accumulators.
__global__ void gemm_nt_kernel(const uint16_t* __restrict__ A,
                               const uint16_t* __restrict__ B,
                               void* __restrict__ C,
                               int M, int N, int K,
                               int lda, int ldb, int ldc,
                               int out_bf16, int store_trans) {
  const int wave = (blockIdx.x * blockDim.x + threadIdx.x) >> 5;
  const int ntiles = N >> 6;
  const int mt = wave / ntiles;
  const int nt = wave - mt * ntiles;
  if (mt >= (M >> 4)) return;
  const int m0 = mt << 4;
  const int n0 = nt << 6;
  const int lane = threadIdx.x & 31;

  v8f acc0 = {0.f,0.f,0.f,0.f,0.f,0.f,0.f,0.f};
  v8f acc1 = acc0, acc2 = acc0, acc3 = acc0;

  for (int k0 = 0; k0 < K; k0 += 32) {
    bf16x16 a  = load_frag_g(A, lda, m0, k0);
    bf16x16 b0 = load_frag_g(B, ldb, n0 +  0, k0);
    bf16x16 b1 = load_frag_g(B, ldb, n0 + 16, k0);
    bf16x16 b2 = load_frag_g(B, ldb, n0 + 32, k0);
    bf16x16 b3 = load_frag_g(B, ldb, n0 + 48, k0);
    acc0 = __builtin_amdgcn_wmma_f32_16x16x32_bf16(false, a, false, b0, (short)0, acc0, false, false);
    acc1 = __builtin_amdgcn_wmma_f32_16x16x32_bf16(false, a, false, b1, (short)0, acc1, false, false);
    acc2 = __builtin_amdgcn_wmma_f32_16x16x32_bf16(false, a, false, b2, (short)0, acc2, false, false);
    acc3 = __builtin_amdgcn_wmma_f32_16x16x32_bf16(false, a, false, b3, (short)0, acc3, false, false);
  }

  // C/D layout: lane L, VGPR v -> row = v + 8*(L>>4), col = L&15
  const int cc = lane & 15;
  const int rbase = m0 + ((lane >> 4) << 3);
  float* Cf = (float*)C;
  uint16_t* Cb = (uint16_t*)C;
#pragma unroll
  for (int v = 0; v < 8; ++v) {
    const int r = rbase + v;
    float vals4[4] = {acc0[v], acc1[v], acc2[v], acc3[v]};
#pragma unroll
    for (int j = 0; j < 4; ++j) {
      const int c = n0 + (j << 4) + cc;
      const size_t idx = store_trans ? ((size_t)c * (size_t)ldc + r)
                                     : ((size_t)r * (size_t)ldc + c);
      if (out_bf16) Cb[idx] = f32_to_bf16(vals4[j]);
      else          Cf[idx] = vals4[j];
    }
  }
}

// ---------------------------------------------------------------- attention
// One block (8 waves == 8 heads) per 16-row s-tile. Two-pass gated softmax:
// pass 1: online row max + denominator; pass 2: exact p, P stored to LDS as
// bf16 A-fragments, attn-mean reduced across heads in LDS, O += P*V via WMMA.
__global__ void GlymphaticGatedCrossAttention_attn_kernel(
    const uint16_t* __restrict__ Qb,   // (N_S, 1024)
    const uint16_t* __restrict__ Kb,   // (N_C, 1024)
    const uint16_t* __restrict__ Vt,   // (1024, N_C) = V^T
    const float*    __restrict__ G,    // (N_S, N_C)
    float* __restrict__ Hattn,         // (N_S, 1024)
    float* __restrict__ attn_mean) {   // (N_S, N_C)
  __shared__ __align__(16) uint16_t p_stage[N_HEADS][16 * 32];
  __shared__ float attn_buf[256];

  const int tid  = threadIdx.x;
  const int lane = tid & 31;
  const int h    = tid >> 5;           // head == wave
  const int s0   = blockIdx.x << 4;
  const int hi   = lane >> 4;
  const int cc   = lane & 15;
  const float scale = 0.08838834764831845f;   // 1/sqrt(128)

  const bf16x16 qf0 = load_frag_g(Qb, SLEEP_DIM, s0, h * D_K +  0);
  const bf16x16 qf1 = load_frag_g(Qb, SLEEP_DIM, s0, h * D_K + 32);
  const bf16x16 qf2 = load_frag_g(Qb, SLEEP_DIM, s0, h * D_K + 64);
  const bf16x16 qf3 = load_frag_g(Qb, SLEEP_DIM, s0, h * D_K + 96);

  float m[8], l[8];
#pragma unroll
  for (int v = 0; v < 8; ++v) { m[v] = -3.0e38f; l[v] = 0.f; }

  const v8f zf = {0.f,0.f,0.f,0.f,0.f,0.f,0.f,0.f};

  // ---------------- pass 1: row max + denominator (online)
  for (int c0 = 0; c0 < N_C; c0 += 16) {
    v8f s = zf;
    s = __builtin_amdgcn_wmma_f32_16x16x32_bf16(false, qf0, false, load_frag_g(Kb, SLEEP_DIM, c0, h*D_K +  0), (short)0, s, false, false);
    s = __builtin_amdgcn_wmma_f32_16x16x32_bf16(false, qf1, false, load_frag_g(Kb, SLEEP_DIM, c0, h*D_K + 32), (short)0, s, false, false);
    s = __builtin_amdgcn_wmma_f32_16x16x32_bf16(false, qf2, false, load_frag_g(Kb, SLEEP_DIM, c0, h*D_K + 64), (short)0, s, false, false);
    s = __builtin_amdgcn_wmma_f32_16x16x32_bf16(false, qf3, false, load_frag_g(Kb, SLEEP_DIM, c0, h*D_K + 96), (short)0, s, false, false);
#pragma unroll
    for (int v = 0; v < 8; ++v) {
      const int r = s0 + v + (hi << 3);
      const float g = G[(size_t)r * N_C + c0 + cc];
      const float x = s[v] * scale * g;
      float mx = x;
      mx = fmaxf(mx, __shfl_xor(mx, 1, 32));
      mx = fmaxf(mx, __shfl_xor(mx, 2, 32));
      mx = fmaxf(mx, __shfl_xor(mx, 4, 32));
      mx = fmaxf(mx, __shfl_xor(mx, 8, 32));
      const float mn = fmaxf(m[v], mx);
      float e = __expf(x - mn);
      e += __shfl_xor(e, 1, 32);
      e += __shfl_xor(e, 2, 32);
      e += __shfl_xor(e, 4, 32);
      e += __shfl_xor(e, 8, 32);
      l[v] = l[v] * __expf(m[v] - mn) + e;
      m[v] = mn;
    }
  }

  float rinv[8];
#pragma unroll
  for (int v = 0; v < 8; ++v) rinv[v] = (l[v] > 0.f) ? (1.0f / l[v]) : 0.f;

  v8f o[8];
#pragma unroll
  for (int j = 0; j < 8; ++j) o[j] = zf;

  // ---------------- pass 2: exact probabilities, attn-mean, O += P*V
  for (int c0 = 0; c0 < N_C; c0 += 32) {
#pragma unroll
    for (int half = 0; half < 2; ++half) {
      const int cb = c0 + (half << 4);
      v8f s = zf;
      s = __builtin_amdgcn_wmma_f32_16x16x32_bf16(false, qf0, false, load_frag_g(Kb, SLEEP_DIM, cb, h*D_K +  0), (short)0, s, false, false);
      s = __builtin_amdgcn_wmma_f32_16x16x32_bf16(false, qf1, false, load_frag_g(Kb, SLEEP_DIM, cb, h*D_K + 32), (short)0, s, false, false);
      s = __builtin_amdgcn_wmma_f32_16x16x32_bf16(false, qf2, false, load_frag_g(Kb, SLEEP_DIM, cb, h*D_K + 64), (short)0, s, false, false);
      s = __builtin_amdgcn_wmma_f32_16x16x32_bf16(false, qf3, false, load_frag_g(Kb, SLEEP_DIM, cb, h*D_K + 96), (short)0, s, false, false);

      float pvals[8];
#pragma unroll
      for (int v = 0; v < 8; ++v) {
        const int rloc = v + (hi << 3);
        const int r = s0 + rloc;
        const float g = G[(size_t)r * N_C + cb + cc];
        const float x = s[v] * scale * g;
        const float p = __expf(x - m[v]) * rinv[v];
        pvals[v] = p;
        p_stage[h][rloc * 32 + (half << 4) + cc] = f32_to_bf16(p);
      }

      // cross-head mean of attn into attn_buf (16x16 tile)
      attn_buf[tid] = 0.f;
      __syncthreads();
#pragma unroll
      for (int v = 0; v < 8; ++v)
        atomicAdd(&attn_buf[((v + (hi << 3)) << 4) + cc], pvals[v] * 0.125f);
      __syncthreads();
      {
        const int rr = tid >> 4, c2 = tid & 15;
        attn_mean[(size_t)(s0 + rr) * N_C + cb + c2] = attn_buf[tid];
      }
      __syncthreads();
    }

    // O(16x128) += P(16x32) * V(32x128): B operand from V^T is K-contiguous
    const bf16x16 pa = load_frag_lds(&p_stage[h][0], 32);
#pragma unroll
    for (int j = 0; j < 8; ++j) {
      const bf16x16 bv = load_frag_g(Vt, N_C, h * D_K + (j << 4), c0);
      o[j] = __builtin_amdgcn_wmma_f32_16x16x32_bf16(false, pa, false, bv, (short)0, o[j], false, false);
    }
  }

  // store Hattn (f32)
#pragma unroll
  for (int j = 0; j < 8; ++j) {
#pragma unroll
    for (int v = 0; v < 8; ++v) {
      const int r = s0 + v + (hi << 3);
      Hattn[(size_t)r * SLEEP_DIM + h * D_K + (j << 4) + cc] = o[j][v];
    }
  }
}

// ---------------------------------------------------------------- bias+res+LN
__global__ void bias_residual_ln_kernel(const float* __restrict__ Hnew,
                                        const float* __restrict__ H_S,
                                        const float* __restrict__ bO,
                                        const float* __restrict__ gamma,
                                        const float* __restrict__ beta,
                                        float* __restrict__ out) {
  __shared__ float red[256];
  const int row = blockIdx.x;
  const int tid = threadIdx.x;
  float x[4];
  float sum = 0.f, sq = 0.f;
#pragma unroll
  for (int i = 0; i < 4; ++i) {
    const int c = tid + (i << 8);
    const float v = Hnew[(size_t)row * SLEEP_DIM + c] + bO[c] +
                    H_S[(size_t)row * SLEEP_DIM + c];
    x[i] = v; sum += v; sq += v * v;
  }
  red[tid] = sum; __syncthreads();
  for (int off = 128; off > 0; off >>= 1) {
    if (tid < off) red[tid] += red[tid + off];
    __syncthreads();
  }
  const float mu = red[0] * (1.0f / SLEEP_DIM);
  __syncthreads();
  red[tid] = sq; __syncthreads();
  for (int off = 128; off > 0; off >>= 1) {
    if (tid < off) red[tid] += red[tid + off];
    __syncthreads();
  }
  const float var = red[0] * (1.0f / SLEEP_DIM) - mu * mu;
  const float rstd = rsqrtf(var + 1e-5f);
#pragma unroll
  for (int i = 0; i < 4; ++i) {
    const int c = tid + (i << 8);
    out[(size_t)row * SLEEP_DIM + c] = (x[i] - mu) * rstd * gamma[c] + beta[c];
  }
}

// ---------------------------------------------------------------- launcher
extern "C" void kernel_launch(void* const* d_in, const int* in_sizes, int n_in,
                              void* d_out, int out_size, void* d_ws, size_t ws_size,
                              hipStream_t stream) {
  const float* H_S  = (const float*)d_in[0];
  const float* H_C  = (const float*)d_in[1];
  const float* G    = (const float*)d_in[2];
  const float* W_Q  = (const float*)d_in[3];
  const float* W_K  = (const float*)d_in[4];
  const float* W_V  = (const float*)d_in[5];
  const float* W_O  = (const float*)d_in[6];
  const float* b_O  = (const float*)d_in[7];
  const float* g_ln = (const float*)d_in[8];
  const float* b_ln = (const float*)d_in[9];

  float* out_main  = (float*)d_out;                       // (N_S, 1024)
  float* attn_mean = out_main + (size_t)N_S * SLEEP_DIM;  // (N_S, N_C)

  // workspace layout (bf16 stored as uint16_t)
  char* ws = (char*)d_ws;
  uint16_t* H_Sb  = (uint16_t*)ws;                  ws += (size_t)N_S * SLEEP_DIM * 2;
  uint16_t* H_Cb  = (uint16_t*)ws;                  ws += (size_t)N_C * BRAIN_DIM * 2;
  uint16_t* W_Qb  = (uint16_t*)ws;                  ws += (size_t)SLEEP_DIM * SLEEP_DIM * 2;
  uint16_t* W_Kb  = (uint16_t*)ws;                  ws += (size_t)SLEEP_DIM * BRAIN_DIM * 2;
  uint16_t* W_Vb  = (uint16_t*)ws;                  ws += (size_t)SLEEP_DIM * BRAIN_DIM * 2;
  uint16_t* W_Ob  = (uint16_t*)ws;                  ws += (size_t)SLEEP_DIM * SLEEP_DIM * 2;
  uint16_t* Qb    = (uint16_t*)ws;                  ws += (size_t)N_S * SLEEP_DIM * 2;
  uint16_t* Kb    = (uint16_t*)ws;                  ws += (size_t)N_C * SLEEP_DIM * 2;
  uint16_t* Vt    = (uint16_t*)ws;                  ws += (size_t)SLEEP_DIM * N_C * 2;
  float*    Hattn = (float*)ws;                     ws += (size_t)N_S * SLEEP_DIM * 4;
  uint16_t* Hatb  = (uint16_t*)ws;                  ws += (size_t)N_S * SLEEP_DIM * 2;
  float*    Hnew  = (float*)ws;                     ws += (size_t)N_S * SLEEP_DIM * 4;

  const int CT = 256;
  // f32 -> bf16 conversions
  cvt_f32_bf16_kernel<<<2048, CT, 0, stream>>>(H_S, H_Sb, N_S * SLEEP_DIM);
  cvt_f32_bf16_kernel<<<2048, CT, 0, stream>>>(H_C, H_Cb, N_C * BRAIN_DIM);
  cvt_f32_bf16_kernel<<<1024, CT, 0, stream>>>(W_Q, W_Qb, SLEEP_DIM * SLEEP_DIM);
  cvt_f32_bf16_kernel<<<512,  CT, 0, stream>>>(W_K, W_Kb, SLEEP_DIM * BRAIN_DIM);
  cvt_f32_bf16_kernel<<<512,  CT, 0, stream>>>(W_V, W_Vb, SLEEP_DIM * BRAIN_DIM);
  cvt_f32_bf16_kernel<<<1024, CT, 0, stream>>>(W_O, W_Ob, SLEEP_DIM * SLEEP_DIM);

  // Q = H_S * W_Q^T            (2048x1024, K=1024) -> bf16
  gemm_nt_kernel<<<(N_S/16)*(SLEEP_DIM/64)/8, CT, 0, stream>>>(
      H_Sb, W_Qb, Qb, N_S, SLEEP_DIM, SLEEP_DIM, SLEEP_DIM, SLEEP_DIM, SLEEP_DIM, 1, 0);
  // K = H_C * W_K^T            (4096x1024, K=512) -> bf16
  gemm_nt_kernel<<<(N_C/16)*(SLEEP_DIM/64)/8, CT, 0, stream>>>(
      H_Cb, W_Kb, Kb, N_C, SLEEP_DIM, BRAIN_DIM, BRAIN_DIM, BRAIN_DIM, SLEEP_DIM, 1, 0);
  // V^T = (H_C * W_V^T)^T      stored transposed (1024 x 4096) -> bf16
  gemm_nt_kernel<<<(N_C/16)*(SLEEP_DIM/64)/8, CT, 0, stream>>>(
      H_Cb, W_Vb, Vt, N_C, SLEEP_DIM, BRAIN_DIM, BRAIN_DIM, BRAIN_DIM, N_C, 1, 1);

  // gated cross-attention (softmax over keys) + attn head-mean
  GlymphaticGatedCrossAttention_attn_kernel<<<N_S/16, CT, 0, stream>>>(
      Qb, Kb, Vt, G, Hattn, attn_mean);

  // O-projection
  cvt_f32_bf16_kernel<<<2048, CT, 0, stream>>>(Hattn, Hatb, N_S * SLEEP_DIM);
  gemm_nt_kernel<<<(N_S/16)*(SLEEP_DIM/64)/8, CT, 0, stream>>>(
      Hatb, W_Ob, Hnew, N_S, SLEEP_DIM, SLEEP_DIM, SLEEP_DIM, SLEEP_DIM, SLEEP_DIM, 0, 0);

  // bias + residual + LayerNorm
  bias_residual_ln_kernel<<<N_S, CT, 0, stream>>>(Hnew, H_S, b_O, g_ln, b_ln, out_main);
}
